// PaLM_9706626089253
// MI455X (gfx1250) — compile-verified, matrix operand order checked by reading, not run on hardware
//
#include <hip/hip_runtime.h>
#include <hip/hip_bf16.h>
#include <math.h>

// ---------------------------------------------------------------------------
// PaLM forward pass for gfx1250 (MI455X).
//  - All GEMMs + attention on V_WMMA_F32_16X16X32_BF16 (wave32).
//  - Weights pre-converted once per GEMM to transposed bf16 [N][K];
//    activations bf16 end-to-end (residual stream f32).
//  - GEMM tiles staged into LDS with GLOBAL_LOAD_ASYNC_TO_LDS_B128
//    (ASYNCcnt path, inline asm) -- raw 16B copies, no conversion in loop.
// ---------------------------------------------------------------------------

#define Bb 2
#define Ss 2048
#define Vv 32000
#define Dd 1024
#define Hh 16
#define DHh 64
#define FFf 4096
#define Mm (Bb * Ss)          // 4096 token rows

typedef __attribute__((ext_vector_type(16))) __bf16    v16bf;
typedef __attribute__((ext_vector_type(8)))  float     v8f;
typedef __attribute__((ext_vector_type(8)))  unsigned  v8u;

static __device__ __forceinline__ unsigned pack_bf16_2(float lo, float hi) {
  unsigned short ul = __builtin_bit_cast(unsigned short, (__bf16)lo);
  unsigned short uh = __builtin_bit_cast(unsigned short, (__bf16)hi);
  return (unsigned)ul | ((unsigned)uh << 16);
}
static __device__ __forceinline__ float bf16lo(unsigned u) {
  return __builtin_bit_cast(float, u << 16);
}
static __device__ __forceinline__ float bf16hi(unsigned u) {
  return __builtin_bit_cast(float, u & 0xffff0000u);
}
static __device__ __forceinline__ v16bf frag(uint4 lo, uint4 hi) {
  const v8u u = {lo.x, lo.y, lo.z, lo.w, hi.x, hi.y, hi.z, hi.w};
  return __builtin_bit_cast(v16bf, u);
}

// LDS byte offset of a generic pointer that points into LDS (AS3 is 32-bit).
typedef __attribute__((address_space(3))) unsigned char lds_byte_t;
static __device__ __forceinline__ unsigned lds_off(void* p) {
  return (unsigned)(unsigned long long)(lds_byte_t*)p;
}
// Async DMA: copy 16 bytes global -> LDS (tracked by ASYNCcnt).
static __device__ __forceinline__ void async_copy_b128(unsigned lds_addr,
                                                       const void* gsrc) {
  asm volatile("global_load_async_to_lds_b128 %0, %1, off"
               :: "v"(lds_addr), "v"(gsrc) : "memory");
}
static __device__ __forceinline__ void wait_async0() {
  asm volatile("s_wait_asynccnt 0" ::: "memory");
}

// ---------------------------------------------------------------------------
// 1) Embedding gather (f32 residual stream)
// ---------------------------------------------------------------------------
__global__ void embed_kernel(const int* __restrict__ x,
                             const float* __restrict__ emb,
                             float* __restrict__ h) {
  const int row = blockIdx.x;
  const int tok = x[row];
  const float* src = emb + (size_t)tok * Dd;
  float* dst = h + (size_t)row * Dd;
  for (int d = threadIdx.x; d < Dd; d += blockDim.x) dst[d] = src[d];
}

// ---------------------------------------------------------------------------
// 2) RMSNorm: f32 in -> bf16 out.  y = rsqrt(sum(x^2)+eps)*x*gamma*sqrt(D)
// ---------------------------------------------------------------------------
__global__ void rmsnorm_kernel(const float* __restrict__ hin,
                               const float* __restrict__ gamma,
                               __bf16* __restrict__ yout) {
  __shared__ float red[256];
  const int row = blockIdx.x;
  const float* xr = hin + (size_t)row * Dd;
  float ss = 0.f;
  for (int d = threadIdx.x; d < Dd; d += 256) {
    float v = xr[d];
    ss += v * v;
  }
  red[threadIdx.x] = ss;
  __syncthreads();
  for (int s = 128; s > 0; s >>= 1) {
    if (threadIdx.x < s) red[threadIdx.x] += red[threadIdx.x + s];
    __syncthreads();
  }
  const float inv = rsqrtf(red[0] + 1e-5f) * sqrtf((float)Dd);
  __bf16* yr = yout + (size_t)row * Dd;
  for (int d = threadIdx.x; d < Dd; d += 256)
    yr[d] = (__bf16)(inv * xr[d] * gamma[d]);
}

// ---------------------------------------------------------------------------
// 3a) Weight conversion: W[K][N] f32 -> Wt[N][K] bf16 (32x32 LDS transpose)
// ---------------------------------------------------------------------------
__global__ __launch_bounds__(256)
void convert_transpose_kernel(const float* __restrict__ in,
                              __bf16* __restrict__ out, int K, int N) {
  __shared__ float tile[32][33];
  const int n0 = blockIdx.x * 32, k0 = blockIdx.y * 32;
  const int tx = threadIdx.x & 31, ty = threadIdx.x >> 5;   // 32 x 8
#pragma unroll
  for (int r = ty; r < 32; r += 8)
    tile[r][tx] = in[(size_t)(k0 + r) * N + n0 + tx];
  __syncthreads();
#pragma unroll
  for (int r = ty; r < 32; r += 8)
    out[(size_t)(n0 + r) * K + k0 + tx] = (__bf16)tile[tx][r];
}

// 3b) Flat f32 -> bf16 (embedding is already [V][D] = [N][K])
__global__ void convert_flat_kernel(const float* __restrict__ in,
                                    unsigned* __restrict__ out, size_t ndw) {
  for (size_t i = blockIdx.x * (size_t)blockDim.x + threadIdx.x; i < ndw;
       i += (size_t)gridDim.x * blockDim.x)
    out[i] = pack_bf16_2(in[2 * i], in[2 * i + 1]);
}

// ---------------------------------------------------------------------------
// 4) WMMA bf16 GEMM:  C[M,N] = A[M,K] x Wt[N,K]^T  (+ residual)
//    A bf16 row-major, Wt bf16 [N][K].  128x64 block tile, 8 waves (4x2),
//    each wave 32x32 => 4 WMMAs/K-step.  Tiles staged via async-to-LDS DMA.
//    outMode: 0 = f32, 1 = f32 + residual, 2 = bf16.
// ---------------------------------------------------------------------------
__global__ __launch_bounds__(256)
void wmma_gemm_kernel(const __bf16* __restrict__ A,
                      const __bf16* __restrict__ W,
                      const float* __restrict__ residual,
                      void* __restrict__ Cout,
                      int M, int N, int K, int outMode) {
  __shared__ alignas(16) unsigned Apk[128][20];  // [m][kpair] 16 used + pad
  __shared__ alignas(16) unsigned Bpk[64][20];   // [n][kpair]

  const int tid  = threadIdx.x;
  const int lane = tid & 31;
  const int wave = tid >> 5;
  const int wm   = wave & 3;
  const int wn   = wave >> 2;

  const int rowBase = blockIdx.y * 128;
  const int colBase = blockIdx.x * 64;

  const int kg = lane >> 4;
  const int lm = lane & 15;

  v8f acc00 = {}, acc01 = {}, acc10 = {}, acc11 = {};

  for (int k0 = 0; k0 < K; k0 += 32) {
    // ---- A tile: 128 rows x 64B -> 512 16B chunks (2 per thread)
#pragma unroll
    for (int it = 0; it < 2; ++it) {
      const int c  = tid + it * 256;        // 0..511
      const int m  = c >> 2, cc = c & 3;
      const char* g = (const char*)A +
                      (((size_t)(rowBase + m) * K + k0) << 1) + cc * 16;
      async_copy_b128(lds_off(&Apk[m][cc * 4]), g);
    }
    // ---- W tile: 64 rows x 64B -> 256 16B chunks (1 per thread)
    {
      const int n = tid >> 2, cc = tid & 3;
      const char* g = (const char*)W +
                      (((size_t)(colBase + n) * K + k0) << 1) + cc * 16;
      async_copy_b128(lds_off(&Bpk[n][cc * 4]), g);
    }
    if (k0 + 32 < K) {
      __builtin_prefetch((const char*)A +
                         (((size_t)(rowBase + (tid >> 1)) * K + k0 + 32) << 1),
                         0, 1);
      __builtin_prefetch((const char*)W +
                         (((size_t)(colBase + (tid & 63)) * K + k0 + 32) << 1),
                         0, 1);
    }
    wait_async0();
    __syncthreads();

    const int am0 = wm * 32 + lm;
    const int am1 = am0 + 16;
    const int bn0 = wn * 32 + lm;
    const int bn1 = bn0 + 16;
    const int pLo = kg * 4;
    const int pHi = 8 + kg * 4;

    const v16bf a0 = frag(*(const uint4*)&Apk[am0][pLo],
                          *(const uint4*)&Apk[am0][pHi]);
    const v16bf a1 = frag(*(const uint4*)&Apk[am1][pLo],
                          *(const uint4*)&Apk[am1][pHi]);
    const v16bf b0 = frag(*(const uint4*)&Bpk[bn0][pLo],
                          *(const uint4*)&Bpk[bn0][pHi]);
    const v16bf b1 = frag(*(const uint4*)&Bpk[bn1][pLo],
                          *(const uint4*)&Bpk[bn1][pHi]);

    acc00 = __builtin_amdgcn_wmma_f32_16x16x32_bf16(
        false, a0, false, b0, (short)0, acc00, false, false);
    acc01 = __builtin_amdgcn_wmma_f32_16x16x32_bf16(
        false, a0, false, b1, (short)0, acc01, false, false);
    acc10 = __builtin_amdgcn_wmma_f32_16x16x32_bf16(
        false, a1, false, b0, (short)0, acc10, false, false);
    acc11 = __builtin_amdgcn_wmma_f32_16x16x32_bf16(
        false, a1, false, b1, (short)0, acc11, false, false);

    __syncthreads();
  }

  // ---- epilogue.  C layout: elem(v,lane): m = v + 8*(lane>>4), n = lane&15
  const int mB = rowBase + wm * 32 + 8 * kg;
  const int nB = colBase + wn * 32 + lm;
#pragma unroll
  for (int v = 0; v < 8; ++v) {
    const size_t r0 = (size_t)(mB + v) * N + nB;
    const size_t r1 = (size_t)(mB + 16 + v) * N + nB;
    if (outMode == 2) {
      __bf16* Cb = (__bf16*)Cout;
      Cb[r0]      = (__bf16)acc00[v];
      Cb[r0 + 16] = (__bf16)acc01[v];
      Cb[r1]      = (__bf16)acc10[v];
      Cb[r1 + 16] = (__bf16)acc11[v];
    } else if (outMode == 1) {
      float* C = (float*)Cout;
      C[r0]      = acc00[v] + residual[r0];
      C[r0 + 16] = acc01[v] + residual[r0 + 16];
      C[r1]      = acc10[v] + residual[r1];
      C[r1 + 16] = acc11[v] + residual[r1 + 16];
    } else {
      float* C = (float*)Cout;
      C[r0]      = acc00[v];
      C[r0 + 16] = acc01[v];
      C[r1]      = acc10[v];
      C[r1 + 16] = acc11[v];
    }
  }
}

// ---------------------------------------------------------------------------
// 5) WMMA attention (bf16 q/k/v/o).  One block per (b, h, 16-query tile).
// ---------------------------------------------------------------------------
#define SPK_STRIDE 1040
#define ATT_LDS_DW (16 * 36 + 16 * SPK_STRIDE + 2 * 64 * 20 + 4 * 16 * 17 + \
                    8 * 16 * 17 + 256 + 16)

__global__ __launch_bounds__(256)
void wmma_attention_kernel(const __bf16* __restrict__ q,
                           const __bf16* __restrict__ k,
                           const __bf16* __restrict__ v,
                           const float* __restrict__ bias,
                           __bf16* __restrict__ o) {
  extern __shared__ unsigned smem[];
  unsigned* Qpk = smem;                          // [16][36]
  unsigned* Spk = Qpk + 16 * 36;                 // [16][1040] packed scores
  unsigned* Vpk = Spk + 16 * SPK_STRIDE;         // [2][64][20]
  float*    Of  = (float*)(Vpk + 2 * 64 * 20);   // [4][16][17]
  float*    Sf  = Of + 4 * 16 * 17;              // [8][16][17]
  float*    red = Sf + 8 * 16 * 17;              // [256]
  float*    invs = red + 256;                    // [16]

  const int tid  = threadIdx.x;
  const int lane = tid & 31;
  const int wave = tid >> 5;
  const int kg   = lane >> 4;
  const int lm   = lane & 15;

  const int ib = blockIdx.x & 127;
  const int hh = (blockIdx.x >> 7) & (Hh - 1);
  const int b  = blockIdx.x >> 11;
  const int i0 = ib * 16;

  const int TK     = ib + 1;
  const int TK32   = (TK + 1) & ~1;
  const int ksteps = TK32 >> 1;
  const float scale = rsqrtf((float)DHh);

  // ---- stage Q (16 x 64 bf16 = 512 dwords), scaled
#pragma unroll
  for (int it = 0; it < 2; ++it) {
    const int t = tid + it * 256;                // 0..511
    const int m = t >> 5, kp = t & 31;
    const unsigned u = ((const unsigned*)(
        q + ((size_t)(b * Ss + i0 + m) * Hh + hh) * DHh))[kp];
    Qpk[m * 36 + kp] = pack_bf16_2(bf16lo(u) * scale, bf16hi(u) * scale);
  }
  __syncthreads();

  const v16bf qf0 = frag(*(const uint4*)&Qpk[lm * 36 + kg * 4],
                         *(const uint4*)&Qpk[lm * 36 + 8 + kg * 4]);
  const v16bf qf1 = frag(*(const uint4*)&Qpk[lm * 36 + 16 + kg * 4],
                         *(const uint4*)&Qpk[lm * 36 + 24 + kg * 4]);

  // ---- phase 1: score tiles
  const int tiles8 = (TK + 7) & ~7;
  for (int jj = 0; jj < tiles8; jj += 8) {
    const int j = jj + wave;                     // wave-uniform
    if (j < TK) {
      const int key = j * 16 + lm;
      // K row is 64 bf16 = 8 uint4; fragment = 4 x 16B global loads
      const uint4* kd = (const uint4*)(k + (size_t)(b * Ss + key) * DHh);
      const v16bf kf0 = frag(kd[kg],     kd[2 + kg]);
      const v16bf kf1 = frag(kd[4 + kg], kd[6 + kg]);
      v8f s = {};
      s = __builtin_amdgcn_wmma_f32_16x16x32_bf16(
          false, qf0, false, kf0, (short)0, s, false, false);
      s = __builtin_amdgcn_wmma_f32_16x16x32_bf16(
          false, qf1, false, kf1, (short)0, s, false, false);

      float* sf = Sf + wave * 16 * 17;
#pragma unroll
      for (int vv = 0; vv < 8; ++vv) {
        const int m = vv + 8 * kg;
        sf[m * 17 + lm] =
            s[vv] + bias[((size_t)hh * Ss + i0 + m) * Ss + j * 16 + lm];
      }
#pragma unroll
      for (int qi = 0; qi < 4; ++qi) {
        const int idx = lane * 4 + qi;
        const int row = idx >> 3, pr = idx & 7;
        Spk[row * SPK_STRIDE + j * 8 + pr] =
            pack_bf16_2(sf[row * 17 + 2 * pr], sf[row * 17 + 2 * pr + 1]);
      }
    }
  }
  if (TK32 > TK && tid < 128) {
    const int row = tid >> 3, pr = tid & 7;
    Spk[row * SPK_STRIDE + TK * 8 + pr] = pack_bf16_2(-1e10f, -1e10f);
  }
  __syncthreads();

  // ---- phase 2: softmax (16 workers per row), bf16 in place
  const int srow = tid >> 4, w = tid & 15;
  const int L = TK32 * 8;
  float lmax = -3.4e38f;
  for (int kp = w; kp < L; kp += 16) {
    const unsigned u = Spk[srow * SPK_STRIDE + kp];
    lmax = fmaxf(lmax, fmaxf(bf16lo(u), bf16hi(u)));
  }
  red[srow * 16 + w] = lmax;
  __syncthreads();
  float rmax = -3.4e38f;
#pragma unroll
  for (int t2 = 0; t2 < 16; ++t2) rmax = fmaxf(rmax, red[srow * 16 + t2]);
  __syncthreads();
  float lsum = 0.f;
  for (int kp = w; kp < L; kp += 16) {
    const unsigned u = Spk[srow * SPK_STRIDE + kp];
    const float plo = __expf(bf16lo(u) - rmax);
    const float phi = __expf(bf16hi(u) - rmax);
    lsum += plo + phi;
    Spk[srow * SPK_STRIDE + kp] = pack_bf16_2(plo, phi);
  }
  red[srow * 16 + w] = lsum;
  __syncthreads();
  if (w == 0) {
    float s2 = 0.f;
#pragma unroll
    for (int t2 = 0; t2 < 16; ++t2) s2 += red[srow * 16 + t2];
    invs[srow] = 1.f / s2;
  }
  __syncthreads();

  // ---- phase 3: O = P.V
  const int nt = wave & 3, par = wave >> 2;
  v8f oacc = {};
  const int nss = (ksteps + 1) >> 1;
  for (int ss = 0; ss < nss; ++ss) {
#pragma unroll
    for (int it = 0; it < 8; ++it) {
      const int t = tid + it * 256;              // 0..2047
      const int chunk = t >> 10;
      const int n  = (t >> 4) & 63;
      const int kp = t & 15;
      const int s  = 2 * ss + chunk;
      if (s < ksteps) {
        const int key = s * 32 + 2 * kp;
        const unsigned short* vr =
            (const unsigned short*)(v + (size_t)(b * Ss + key) * DHh);
        const unsigned lo = vr[n];
        const unsigned hi = vr[DHh + n];         // next key row
        Vpk[(chunk * 64 + n) * 20 + kp] = lo | (hi << 16);
      }
    }
    __syncthreads();
    const int s = 2 * ss + par;                  // wave-uniform
    if (s < ksteps) {
      const v16bf pf =
          frag(*(const uint4*)&Spk[lm * SPK_STRIDE + 16 * s + kg * 4],
               *(const uint4*)&Spk[lm * SPK_STRIDE + 16 * s + 8 + kg * 4]);
      const unsigned* vb = &Vpk[(par * 64 + nt * 16 + lm) * 20];
      const v16bf vf = frag(*(const uint4*)&vb[kg * 4],
                            *(const uint4*)&vb[8 + kg * 4]);
      oacc = __builtin_amdgcn_wmma_f32_16x16x32_bf16(
          false, pf, false, vf, (short)0, oacc, false, false);
    }
    __syncthreads();
  }

  if (par == 1) {
#pragma unroll
    for (int vv = 0; vv < 8; ++vv)
      Of[(nt * 16 + vv + 8 * kg) * 17 + lm] = oacc[vv];
  }
  __syncthreads();
  if (par == 0) {
#pragma unroll
    for (int vv = 0; vv < 8; ++vv) {
      const int m = vv + 8 * kg;
      const float val = (oacc[vv] + Of[(nt * 16 + m) * 17 + lm]) * invs[m];
      o[((size_t)(b * Ss + i0 + m) * Hh + hh) * DHh + nt * 16 + lm] =
          (__bf16)val;
    }
  }
}

// ---------------------------------------------------------------------------
// 6) SwiGLU elementwise on packed bf16:  u <- u * (g * sigmoid(g))
// ---------------------------------------------------------------------------
__global__ void swiglu_kernel(unsigned* __restrict__ u,
                              const unsigned* __restrict__ g, size_t ndw) {
  for (size_t i = blockIdx.x * (size_t)blockDim.x + threadIdx.x; i < ndw;
       i += (size_t)gridDim.x * blockDim.x) {
    const unsigned uu = u[i], gg = g[i];
    const float g0 = bf16lo(gg), g1 = bf16hi(gg);
    const float r0 = bf16lo(uu) * g0 / (1.f + __expf(-g0));
    const float r1 = bf16hi(uu) * g1 / (1.f + __expf(-g1));
    u[i] = pack_bf16_2(r0, r1);
  }
}

// ---------------------------------------------------------------------------
// Host-side orchestration
// ---------------------------------------------------------------------------
extern "C" void kernel_launch(void* const* d_in, const int* in_sizes, int n_in,
                              void* d_out, int out_size, void* d_ws,
                              size_t ws_size, hipStream_t stream) {
  (void)in_sizes; (void)n_in; (void)out_size; (void)ws_size;

  const int*   x          = (const int*)  d_in[0];
  const float* embedding  = (const float*)d_in[1];
  const float* attn_bias  = (const float*)d_in[2];
  const float* attn_gamma = (const float*)d_in[3];
  const float* wq         = (const float*)d_in[4];
  const float* wk         = (const float*)d_in[5];
  const float* wv         = (const float*)d_in[6];
  const float* wo_attn    = (const float*)d_in[7];
  const float* ff_gamma   = (const float*)d_in[8];
  const float* wi         = (const float*)d_in[9];
  const float* wg         = (const float*)d_in[10];
  const float* wo_ff      = (const float*)d_in[11];
  const float* fgamma     = (const float*)d_in[12];
  float* out = (float*)d_out;

  char* ws = (char*)d_ws;
  size_t off = 0;
  auto carveB = [&](size_t bytes) {
    void* p = ws + off;
    off += (bytes + 15) & ~(size_t)15;
    return p;
  };
  float*  h    = (float*) carveB((size_t)Mm * Dd * 4);       // residual (f32)
  __bf16* y    = (__bf16*)carveB((size_t)Mm * Dd * 2);
  __bf16* qb   = (__bf16*)carveB((size_t)Mm * Hh * DHh * 2);
  __bf16* kb   = (__bf16*)carveB((size_t)Mm * DHh * 2);
  __bf16* vb   = (__bf16*)carveB((size_t)Mm * DHh * 2);
  __bf16* ob   = (__bf16*)carveB((size_t)Mm * Hh * DHh * 2);
  __bf16* ub   = (__bf16*)carveB((size_t)Mm * FFf * 2);
  __bf16* gb   = (__bf16*)carveB((size_t)Mm * FFf * 2);
  __bf16* wt   = (__bf16*)carveB((size_t)FFf * Dd * 2);      // weight scratch
  __bf16* embB = (__bf16*)carveB((size_t)Vv * Dd * 2);       // bf16 embedding

  const dim3 blk(256);

  auto convT = [&](const float* w, int K, int N) {   // [K][N] f32 -> [N][K] bf16
    convert_transpose_kernel<<<dim3(N / 32, K / 32), blk, 0, stream>>>(w, wt,
                                                                       K, N);
  };
  auto gemm = [&](const __bf16* A, const __bf16* W, const float* res,
                  void* C, int M, int N, int K, int mode) {
    dim3 grid(N / 64, M / 128);
    wmma_gemm_kernel<<<grid, blk, 0, stream>>>(A, W, res, C, M, N, K, mode);
  };

  embed_kernel<<<dim3(Mm), blk, 0, stream>>>(x, embedding, h);
  convert_flat_kernel<<<dim3(2048), blk, 0, stream>>>(
      embedding, (unsigned*)embB, (size_t)Vv * Dd / 2);

  for (int l = 0; l < 8; ++l) {
    const float* aG  = attn_gamma + (size_t)l * Dd;
    const float* aWq = wq      + (size_t)l * Dd * (Hh * DHh);
    const float* aWk = wk      + (size_t)l * Dd * DHh;
    const float* aWv = wv      + (size_t)l * Dd * DHh;
    const float* aWo = wo_attn + (size_t)l * (Hh * DHh) * Dd;
    const float* fG  = ff_gamma + (size_t)l * Dd;
    const float* fWi = wi    + (size_t)l * Dd * FFf;
    const float* fWg = wg    + (size_t)l * Dd * FFf;
    const float* fWo = wo_ff + (size_t)l * FFf * Dd;

    // ---- attention block
    rmsnorm_kernel<<<dim3(Mm), blk, 0, stream>>>(h, aG, y);
    convT(aWq, Dd, Hh * DHh);
    gemm(y, wt, nullptr, qb, Mm, Hh * DHh, Dd, 2);
    convT(aWk, Dd, DHh);
    gemm(y, wt, nullptr, kb, Mm, DHh, Dd, 2);
    convT(aWv, Dd, DHh);
    gemm(y, wt, nullptr, vb, Mm, DHh, Dd, 2);
    wmma_attention_kernel<<<dim3(Bb * Hh * (Ss / 16)), blk,
                            ATT_LDS_DW * sizeof(unsigned), stream>>>(
        qb, kb, vb, attn_bias, ob);
    convT(aWo, Hh * DHh, Dd);
    gemm(ob, wt, h, h, Mm, Dd, Hh * DHh, 1);    // h += o @ Wo

    // ---- feed-forward block (SwiGLU)
    rmsnorm_kernel<<<dim3(Mm), blk, 0, stream>>>(h, fG, y);
    convT(fWi, Dd, FFf);
    gemm(y, wt, nullptr, ub, Mm, FFf, Dd, 2);
    convT(fWg, Dd, FFf);
    gemm(y, wt, nullptr, gb, Mm, FFf, Dd, 2);
    swiglu_kernel<<<dim3(2048), blk, 0, stream>>>(
        (unsigned*)ub, (const unsigned*)gb, (size_t)Mm * FFf / 2);
    convT(fWo, FFf, Dd);
    gemm(ub, wt, h, h, Mm, Dd, FFf, 1);         // h += ffn @ Wo_ff
  }

  // final norm + logits (embB is already [V][D] = [N][K])
  rmsnorm_kernel<<<dim3(Mm), blk, 0, stream>>>(h, fgamma, y);
  gemm(y, embB, nullptr, out, Mm, Vv, Dd, 0);
}
